// CausalSelfAttention_18425409700013
// MI455X (gfx1250) — compile-verified, matrix operand order checked by reading, not run on hardware
//
#include <hip/hip_runtime.h>
#include <stdint.h>
#include <stddef.h>

// Problem constants (CausalSelfAttention: B=1, S=4096, E=1024, H=16, D=64)
#define S_LEN 4096
#define EMB   1024
#define NHEAD 16
#define HDIM  64
#define QKV_N 3072

typedef __bf16 bf16_t;
typedef bf16_t v8bf  __attribute__((ext_vector_type(8)));
typedef bf16_t v16bf __attribute__((ext_vector_type(16)));
typedef float  v8f   __attribute__((ext_vector_type(8)));
typedef float  v4f   __attribute__((ext_vector_type(4)));
typedef uint32_t u32x4 __attribute__((ext_vector_type(4)));
typedef int      i32x4 __attribute__((ext_vector_type(4)));
typedef int      i32x8 __attribute__((ext_vector_type(8)));

union Frag16 { v16bf v; v8bf h[2]; };

__device__ __forceinline__ v8f v8f_zero() {
  v8f v;
#pragma unroll
  for (int i = 0; i < 8; ++i) v[i] = 0.0f;
  return v;
}

// Async copy: 16 bytes global -> LDS, tracked by ASYNCcnt (no VGPR data path).
__device__ __forceinline__ void async_b128(uint32_t lds_off, uint32_t goff,
                                           const void* base) {
  uint64_t b = (uint64_t)base;
  asm volatile("global_load_async_to_lds_b128 %0, %1, %2"
               :
               : "v"(lds_off), "v"(goff), "s"(b)
               : "memory");
}
__device__ __forceinline__ void wait_async0() {
  asm volatile("s_wait_asynccnt 0x0" ::: "memory");
}
// Flat __shared__ pointer -> 32-bit LDS byte offset (low 32 bits of aperture addr).
__device__ __forceinline__ uint32_t lds_off(const void* p) {
  return (uint32_t)(uintptr_t)p;
}

// Tensor Data Mover: DMA a 2D bf16 tile (rows x cols, row stride in elements)
// from global memory into LDS, padding each row of cols*2 bytes with
// pad_amount DWORDs every pad_interval DWORDs (codes per D# group1 encoding).
// One instruction per tile, tracked by TENSORcnt. Must be issued wave-uniform.
__device__ __forceinline__ void tdm_load_2d(uint32_t lds_byte_off,
                                            const void* gtile, uint32_t rows,
                                            uint32_t cols, uint32_t row_stride,
                                            uint32_t pad_int_code,
                                            uint32_t pad_amt_code) {
  const uint64_t ga = (uint64_t)gtile;
  u32x4 g0;
  g0[0] = 1u;                        // count=1 valid descriptor, user mode
  g0[1] = lds_byte_off;              // lds_addr [63:32]
  g0[2] = (uint32_t)ga;              // global_addr [95:64]
  g0[3] = (uint32_t)((ga >> 32) & 0x01FFFFFFu) | (2u << 30);  // addr[56:32], type=2
  i32x8 g1;
  g1[0] = (int)((1u << 16) |                    // data_size = 2 bytes
                (1u << 20) |                    // pad_enable
                (pad_int_code << 22) |          // pad_interval
                (pad_amt_code << 25));          // pad_amount
  g1[1] = (int)((cols & 0xFFFFu) << 16);        // tensor_dim0[15:0]
  g1[2] = (int)((rows & 0xFFFFu) << 16);        // tensor_dim0[31:16]=0 | tensor_dim1[15:0]
  g1[3] = (int)((cols & 0xFFFFu) << 16);        // tensor_dim1[31:16]=0 | tile_dim0
  g1[4] = (int)(rows & 0xFFFFu);                // tile_dim1 | tile_dim2=0
  g1[5] = (int)row_stride;                      // tensor_dim0_stride[31:0]
  g1[6] = 0;                                    // stride hi | dim1_stride lo
  g1[7] = 0;                                    // dim1_stride hi
  i32x4 z4 = {0, 0, 0, 0};
  i32x8 z8 = {0, 0, 0, 0, 0, 0, 0, 0};
  // 6-arg form (amdgpu-toolchain clang-23): (g0, g1, g2, g3, g4, cpol)
  __builtin_amdgcn_tensor_load_to_lds(g0, g1, z4, z4, z8, 0);
}

// Load a 16x32 bf16 WMMA A/B fragment from LDS (row-major rows of `stride`
// bf16 elements). CDNA5 16-bit operand layout: lane L holds row (L&15);
// lanes 0-15 cover K={0..7,16..23}, lanes 16-31 cover K={8..15,24..31}.
__device__ __forceinline__ v16bf ld_frag(const bf16_t* base, int stride, int lane) {
  const bf16_t* p = base + (lane & 15) * stride + ((lane >> 4) << 3);
  Frag16 f;
  f.h[0] = *(const v8bf*)(p);
  f.h[1] = *(const v8bf*)(p + 16);
  return f.v;
}

__device__ __forceinline__ v8f wmma_bf16(v16bf a, v16bf b, v8f c) {
  return __builtin_amdgcn_wmma_f32_16x16x32_bf16(false, a, false, b, (short)0, c,
                                                 false, false);
}

// ---------------------------------------------------------------------------
// Pre-pass: f32 -> bf16 elementwise (x), and f32 -> bf16 transpose (weights),
// so all GEMM/attention staging is pure bf16 row copies (async / TDM path).
// ---------------------------------------------------------------------------
__global__ __launch_bounds__(256) void cvt_bf16(const float* __restrict__ src,
                                                bf16_t* __restrict__ dst, int n) {
  int i = (blockIdx.x * 256 + threadIdx.x) * 4;
  if (i + 3 < n) {
    v4f v = *(const v4f*)(src + i);
    dst[i + 0] = (bf16_t)v.x;
    dst[i + 1] = (bf16_t)v.y;
    dst[i + 2] = (bf16_t)v.z;
    dst[i + 3] = (bf16_t)v.w;
  }
}

// dst[C][R] = (bf16) src[R][C], 32x32 LDS tiles, both sides coalesced.
__global__ __launch_bounds__(256) void transpose_cvt(
    const float* __restrict__ src, bf16_t* __restrict__ dst, int R, int C) {
  __shared__ bf16_t tile[32][33];
  const int bx = blockIdx.x * 32;  // source col base
  const int by = blockIdx.y * 32;  // source row base
  const int tx = threadIdx.x & 31;
  const int ty = threadIdx.x >> 5;  // 0..7
#pragma unroll
  for (int i = 0; i < 32; i += 8)
    tile[ty + i][tx] = (bf16_t)src[(size_t)(by + ty + i) * C + bx + tx];
  __syncthreads();
#pragma unroll
  for (int i = 0; i < 32; i += 8)
    dst[(size_t)(bx + ty + i) * R + by + tx] = tile[tx][ty + i];
}

// ---------------------------------------------------------------------------
// Tiled GEMM: C[M,N] = A[M,K] (bf16) * BT[N,K]^T (bf16) + bias[N]
// 128x128 block tile, 32-wide K chunks, 8 waves (4x2), async-to-LDS staging
// with ping-pong double buffering so the copy engine overlaps the WMMAs.
// ---------------------------------------------------------------------------
template <bool OUT_BF16>
__global__ __launch_bounds__(256) void wmma_gemm(
    const bf16_t* __restrict__ A, const bf16_t* __restrict__ BT,
    const float* __restrict__ bias, void* __restrict__ C, int M, int N, int K,
    int scale_cols, float scale) {
  __shared__ alignas(16) bf16_t As[2][128][40];  // [m][k], padded rows (80B)
  __shared__ alignas(16) bf16_t Bs[2][128][40];  // [n][k]

  const int tid  = threadIdx.x;
  const int lane = tid & 31;
  const int wave = tid >> 5;
  const int wm   = wave >> 1;  // 0..3
  const int wn   = wave & 1;   // 0..1
  const int m0   = blockIdx.y * 128;
  const int n0   = blockIdx.x * 128;

  v8f acc[2][4];
#pragma unroll
  for (int mi = 0; mi < 2; ++mi)
#pragma unroll
    for (int ni = 0; ni < 4; ++ni) acc[mi][ni] = v8f_zero();

  // Stage one 128x32 bf16 tile pair: 512 16B chunks each, 2 per thread.
  auto stage = [&](int buf, int k0) {
#pragma unroll
    for (int rep = 0; rep < 2; ++rep) {
      int idx = rep * 256 + tid;  // 0..511
      int r   = idx >> 2;         // 0..127
      int c8  = (idx & 3) * 8;    // 0..24
      async_b128(lds_off(&As[buf][r][c8]),
                 (uint32_t)(((size_t)(m0 + r) * K + k0 + c8) * sizeof(bf16_t)), A);
      async_b128(lds_off(&Bs[buf][r][c8]),
                 (uint32_t)(((size_t)(n0 + r) * K + k0 + c8) * sizeof(bf16_t)), BT);
    }
  };

  stage(0, 0);
  for (int k0 = 0; k0 < K; k0 += 32) {
    const int buf = (k0 >> 5) & 1;
    wait_async0();    // this wave's in-flight chunk (issued last iter) is in LDS
    __syncthreads();  // all waves' chunks in LDS; prev buffer's readers done
    if (k0 + 32 < K) stage(buf ^ 1, k0 + 32);

    v16bf af[2];
#pragma unroll
    for (int mi = 0; mi < 2; ++mi)
      af[mi] = ld_frag(&As[buf][wm * 32 + mi * 16][0], 40, lane);
    v16bf bfr[4];
#pragma unroll
    for (int ni = 0; ni < 4; ++ni)
      bfr[ni] = ld_frag(&Bs[buf][wn * 64 + ni * 16][0], 40, lane);
#pragma unroll
    for (int mi = 0; mi < 2; ++mi)
#pragma unroll
      for (int ni = 0; ni < 4; ++ni)
        acc[mi][ni] = wmma_bf16(af[mi], bfr[ni], acc[mi][ni]);
  }

  // Epilogue. C/D layout: lane L, vgpr r -> (M = r + 8*(L>>4), N = L&15).
  const int halfsel = lane >> 4;
#pragma unroll
  for (int mi = 0; mi < 2; ++mi) {
    const int mbase = m0 + wm * 32 + mi * 16 + halfsel * 8;
#pragma unroll
    for (int ni = 0; ni < 4; ++ni) {
      const int ng     = n0 + wn * 64 + ni * 16 + (lane & 15);
      const float bv   = bias[ng];
      const float mult = (ng < scale_cols) ? scale : 1.0f;
#pragma unroll
      for (int r = 0; r < 8; ++r) {
        float v = (acc[mi][ni][r] + bv) * mult;
        if constexpr (OUT_BF16)
          ((bf16_t*)C)[(size_t)(mbase + r) * N + ng] = (bf16_t)v;
        else
          ((float*)C)[(size_t)(mbase + r) * N + ng] = v;
      }
    }
  }
}

// ---------------------------------------------------------------------------
// Flash attention (causal). One workgroup per (head, 64-query tile).
// 4 waves, each wave owns 16 query rows. qkv bf16 [S,3E], 1/sqrt(D) folded
// into Q. Q/K tiles DMA'd by the Tensor Data Mover (one tensor_load_to_lds
// per tile, D# pad fields reproduce the 72-element padded LDS rows); V is
// staged transposed manually. Output bf16 [S,E].
// ---------------------------------------------------------------------------
__global__ __launch_bounds__(128) void flash_attn(
    const bf16_t* __restrict__ qkv, bf16_t* __restrict__ attn) {
  __shared__ alignas(16) bf16_t Qs[64][72];     // [q][d]
  __shared__ alignas(16) bf16_t Ks[64][72];     // [key][d]  (= B^T for QK^T)
  __shared__ alignas(16) bf16_t Vts[64][72];    // [d][key]  (= B^T for P*V)
  __shared__ alignas(16) bf16_t Ps[4][16][72];  // per-wave P tile [q][key]

  const int h    = blockIdx.x;
  const int qt   = blockIdx.y;
  const int q0   = qt * 64;
  const int tid  = threadIdx.x;
  const int lane = tid & 31;
  const int wave = tid >> 5;

  const int cq = h * HDIM;
  const int ck = EMB + h * HDIM;
  const int cv = 2 * EMB + h * HDIM;

  // Row = 64 bf16 = 128 B = 32 DWORDs -> pad_interval code 4 (32 DW);
  // pad 16 B = 4 DWORDs -> pad_amount code 3; LDS row stride = 72 bf16.
  if (wave == 0)
    tdm_load_2d(lds_off(&Qs[0][0]), qkv + (size_t)q0 * QKV_N + cq, 64, 64,
                QKV_N, 4, 3);

  v8f acc[4];
#pragma unroll
  for (int ni = 0; ni < 4; ++ni) acc[ni] = v8f_zero();
  v8f mrow, lrow;
#pragma unroll
  for (int r = 0; r < 8; ++r) { mrow[r] = -1e30f; lrow[r] = 0.0f; }

  for (int j = 0; j <= qt; ++j) {
    const int k0 = j * 64;
    __syncthreads();  // previous iteration's K/Vt/Ps readers done
    if (wave == 0)    // K tile via TDM
      tdm_load_2d(lds_off(&Ks[0][0]), qkv + (size_t)k0 * QKV_N + ck, 64, 64,
                  QKV_N, 4, 3);
    // V chunk: manual transpose into Vts (TDM cannot transpose).
#pragma unroll
    for (int rep = 0; rep < 4; ++rep) {
      int idx = rep * 128 + tid;
      int r   = idx >> 3;
      int c8  = (idx & 7) * 8;
      v8bf vv = *(const v8bf*)(qkv + (size_t)(k0 + r) * QKV_N + cv + c8);
#pragma unroll
      for (int t = 0; t < 8; ++t) Vts[c8 + t][r] = vv[t];
    }
    if (wave == 0) __builtin_amdgcn_s_wait_tensorcnt(0);  // Q (1st iter) + K
    __syncthreads();  // publish TDM tiles + V stores to all waves

    // Scores S = Q * K^T  (16 queries x 64 keys per wave).
    v8f sc[4];
#pragma unroll
    for (int ni = 0; ni < 4; ++ni) sc[ni] = v8f_zero();
#pragma unroll
    for (int kc = 0; kc < 2; ++kc) {
      v16bf aq = ld_frag(&Qs[wave * 16][kc * 32], 72, lane);
#pragma unroll
      for (int ni = 0; ni < 4; ++ni) {
        v16bf bk = ld_frag(&Ks[ni * 16][kc * 32], 72, lane);
        sc[ni] = wmma_bf16(aq, bk, sc[ni]);
      }
    }

    // Causal mask only on the diagonal chunk.
    if (j == qt) {
      const int qrow_base = wave * 16 + (lane >> 4) * 8;
#pragma unroll
      for (int ni = 0; ni < 4; ++ni) {
        const int kcol = ni * 16 + (lane & 15);
#pragma unroll
        for (int r = 0; r < 8; ++r)
          if (kcol > qrow_base + r) sc[ni][r] = -1e30f;
      }
    }

    // Row max across 4 n-tiles, then across the 16-lane half-group.
    v8f mt = sc[0];
#pragma unroll
    for (int ni = 1; ni < 4; ++ni)
#pragma unroll
      for (int r = 0; r < 8; ++r) mt[r] = fmaxf(mt[r], sc[ni][r]);
#pragma unroll
    for (int m = 1; m <= 8; m <<= 1)
#pragma unroll
      for (int r = 0; r < 8; ++r) mt[r] = fmaxf(mt[r], __shfl_xor(mt[r], m, 32));

    v8f mnew, fac;
#pragma unroll
    for (int r = 0; r < 8; ++r) {
      mnew[r] = fmaxf(mrow[r], mt[r]);
      fac[r]  = __expf(mrow[r] - mnew[r]);
    }

    // P = exp(S - m_new); spill to LDS in C layout, re-read in A layout.
    v8f rs = v8f_zero();
    const int prow_base = (lane >> 4) * 8;
#pragma unroll
    for (int ni = 0; ni < 4; ++ni) {
      const int kcol = ni * 16 + (lane & 15);
#pragma unroll
      for (int r = 0; r < 8; ++r) {
        float p = __expf(sc[ni][r] - mnew[r]);
        rs[r] += p;
        Ps[wave][prow_base + r][kcol] = (bf16_t)p;
      }
    }
#pragma unroll
    for (int m = 1; m <= 8; m <<= 1)
#pragma unroll
      for (int r = 0; r < 8; ++r) rs[r] += __shfl_xor(rs[r], m, 32);

#pragma unroll
    for (int r = 0; r < 8; ++r) {
      lrow[r] = lrow[r] * fac[r] + rs[r];
      mrow[r] = mnew[r];
    }
#pragma unroll
    for (int ni = 0; ni < 4; ++ni)
#pragma unroll
      for (int r = 0; r < 8; ++r) acc[ni][r] *= fac[r];

    __syncthreads();  // P visible before fragment re-reads

    // O += P * V
#pragma unroll
    for (int kc = 0; kc < 2; ++kc) {
      v16bf ap = ld_frag(&Ps[wave][0][kc * 32], 72, lane);
#pragma unroll
      for (int ni = 0; ni < 4; ++ni) {
        v16bf bv = ld_frag(&Vts[ni * 16][kc * 32], 72, lane);
        acc[ni] = wmma_bf16(ap, bv, acc[ni]);
      }
    }
  }

  // Normalize, store bf16 attention output [S, E].
  const int qbase = q0 + wave * 16 + (lane >> 4) * 8;
#pragma unroll
  for (int ni = 0; ni < 4; ++ni) {
    const int col = h * HDIM + ni * 16 + (lane & 15);
#pragma unroll
    for (int r = 0; r < 8; ++r)
      attn[(size_t)(qbase + r) * EMB + col] = (bf16_t)(acc[ni][r] / lrow[r]);
  }
}

// ---------------------------------------------------------------------------
extern "C" void kernel_launch(void* const* d_in, const int* in_sizes, int n_in,
                              void* d_out, int out_size, void* d_ws,
                              size_t ws_size, hipStream_t stream) {
  const float* x     = (const float*)d_in[0];
  const float* w_qkv = (const float*)d_in[1];
  const float* b_qkv = (const float*)d_in[2];
  const float* w_out = (const float*)d_in[3];
  const float* b_out = (const float*)d_in[4];

  // Workspace layout (bf16):
  //   qkv   [4096,3072]  24 MB
  //   attn  [4096,1024]   8 MB
  //   xb    [4096,1024]   8 MB
  //   wqkvT [3072,1024]   6 MB
  //   woutT [1024,1024]   2 MB
  char* ws = (char*)d_ws;
  bf16_t* qkv   = (bf16_t*)(ws);
  bf16_t* attn  = (bf16_t*)(ws + 25165824);
  bf16_t* xb    = (bf16_t*)(ws + 25165824 + 8388608);
  bf16_t* wqkvT = (bf16_t*)(ws + 25165824 + 8388608 + 8388608);
  bf16_t* woutT = (bf16_t*)(ws + 25165824 + 8388608 + 8388608 + 6291456);

  // 0) One-shot conversions so all hot-loop staging is bf16 copies.
  cvt_bf16<<<(S_LEN * EMB) / 1024, 256, 0, stream>>>(x, xb, S_LEN * EMB);
  {
    dim3 g(QKV_N / 32, EMB / 32);
    transpose_cvt<<<g, 256, 0, stream>>>(w_qkv, wqkvT, EMB, QKV_N);
  }
  {
    dim3 g(EMB / 32, EMB / 32);
    transpose_cvt<<<g, 256, 0, stream>>>(w_out, woutT, EMB, EMB);
  }

  // 1) qkv = x @ w_qkv + b_qkv (bf16 out; 1/sqrt(64) folded into Q columns).
  dim3 g1(QKV_N / 128, S_LEN / 128);
  wmma_gemm<true><<<g1, 256, 0, stream>>>(xb, wqkvT, b_qkv, (void*)qkv, S_LEN,
                                          QKV_N, EMB, EMB, 0.125f);

  // 2) causal flash attention per (head, 64-query tile).
  dim3 g2(NHEAD, S_LEN / 64);
  flash_attn<<<g2, 128, 0, stream>>>(qkv, attn);

  // 3) out = attn @ w_out + b_out (f32 out).
  dim3 g3(EMB / 128, S_LEN / 128);
  wmma_gemm<false><<<g3, 256, 0, stream>>>(attn, woutT, b_out, d_out, S_LEN,
                                           EMB, EMB, 0, 1.0f);
}
